// JointPLE_30897994727931
// MI455X (gfx1250) — matrix operand order repaired and checked by reading, not compiled, not run on hardware
//
#include <hip/hip_runtime.h>
#include <math.h>

// ---------------- model dims ----------------
#define B_  64
#define L_  512
#define H_  768
#define A_  18
#define Dm  1536            // 2*H
#define EPS_LN 1e-5f

// GEMM tiling
#define TM 64               // block tile M
#define TN 128              // block tile N
#define TK 32               // K step (one bf16 WMMA depth)

typedef __attribute__((ext_vector_type(16))) __bf16 v16bf;
typedef __attribute__((ext_vector_type(8)))  float  v8f;

union BF16x16 { unsigned int u[8]; v16bf v; };

__device__ __forceinline__ unsigned int pack_bf16(float a, float b) {
  unsigned int ua = __float_as_uint(a);
  unsigned int ub = __float_as_uint(b);
  unsigned int ra = (ua + 0x7FFFu + ((ua >> 16) & 1u)) >> 16;
  unsigned int rb = (ub + 0x7FFFu + ((ub >> 16) & 1u)) >> 16;
  return (ra & 0xFFFFu) | (rb << 16);
}

__device__ __forceinline__ float gelu_exact(float x) {
  return 0.5f * x * (1.0f + erff(x * 0.70710678118654752f));
}

// ---- fragment gather from packed-pair LDS rows (ISA 7.12.2 layouts) ----
__device__ __forceinline__ void gather_frags(const unsigned int* __restrict__ sAbuf,
                                             const unsigned int* __restrict__ sBbuf,
                                             int wm, int wn, int g, int r,
                                             BF16x16& a0, BF16x16& a1,
                                             BF16x16& b0, BF16x16& b1) {
  const unsigned int* rowA0 = &sAbuf[(wm + r) * 17];
  const unsigned int* rowA1 = &sAbuf[(wm + 16 + r) * 17];
  const unsigned int* rowB0 = &sBbuf[(wn + r) * 17];
  const unsigned int* rowB1 = &sBbuf[(wn + 16 + r) * 17];
  // A 16x32: VGPR v holds K=(v/4)*16 + g*8 + (v%4)*2 -> dword (v/4)*8+g*4+(v%4)
#pragma unroll
  for (int v = 0; v < 4; v++) {
    a0.u[v] = rowA0[g * 4 + v];  a0.u[4 + v] = rowA0[8 + g * 4 + v];
    a1.u[v] = rowA1[g * 4 + v];  a1.u[4 + v] = rowA1[8 + g * 4 + v];
  }
  // B 32x16: VGPR v holds K=(v/4)*16 + g*8 + (v%4)*2 -> dword (v/4)*8+g*4+(v%4)
#pragma unroll
  for (int v = 0; v < 8; v++) { b0.u[v] = rowB0[(v >> 2) * 8 + g * 4 + (v & 3)]; b1.u[v] = rowB1[(v >> 2) * 8 + g * 4 + (v & 3)]; }
}

__device__ __forceinline__ void epilogue_store(v8f& acc00, v8f& acc01, v8f& acc10, v8f& acc11,
                                               const float* biasb, float* Cb, int ldc,
                                               int m0, int n0, int wm, int wn, int g, int r,
                                               int M, int N, int doGelu, int doAccum, float alpha) {
  v8f* accs[4] = { &acc00, &acc01, &acc10, &acc11 };
#pragma unroll
  for (int ti = 0; ti < 4; ti++) {
    int mi = (ti >> 1) * 16, ni = (ti & 1) * 16;
    int nIdx = n0 + wn + ni + r;
    v8f& a = *accs[ti];
#pragma unroll
    for (int rr = 0; rr < 8; rr++) {
      int mIdx = m0 + wm + mi + rr + g * 8;
      if (mIdx < M && nIdx < N) {
        float x = a[rr] * alpha;
        if (biasb) x += biasb[nIdx];
        long long ci = (long long)mIdx * ldc + nIdx;
        if (doAccum) x += Cb[ci];
        if (doGelu) x = gelu_exact(x);
        Cb[ci] = x;
      }
    }
  }
}

// =====================================================================
// FAST GEMM: requires M%64==0, N%128==0, K%32==0, no transposes.
// 64x128 block tile; 8 waves as 2x4; 4 WMMAs/wave/K-step; LDS double-buffer.
// =====================================================================
__global__ __launch_bounds__(256)
void gemm_bf16_fast(const float* __restrict__ Ag, int lda, long long strA,
                    const float* __restrict__ Bg, int ldb, long long strB,
                    const float* __restrict__ bias, long long strBias,
                    float* __restrict__ Cg, int ldc, long long strC,
                    int K, int doGelu, int doAccum, float alpha)
{
  __shared__ unsigned int sA[2][TM * 17];
  __shared__ unsigned int sB[2][TN * 17];

  const float* Ab = Ag + (long long)blockIdx.z * strA;
  const float* Bb = Bg + (long long)blockIdx.z * strB;
  const float* biasb = bias ? (bias + (long long)blockIdx.z * strBias) : nullptr;
  float* Cb = Cg + (long long)blockIdx.z * strC;

  const int m0 = blockIdx.y * TM;
  const int n0 = blockIdx.x * TN;
  const int t = threadIdx.x;
  const int lane = t & 31;
  const int wave = t >> 5;
  const int wm = (wave & 1) * 32;
  const int wn = (wave >> 1) * 32;
  const int g = lane >> 4;
  const int r = lane & 15;

  // thread-invariant staging coordinates
  const int ai = t >> 4, aslot = t & 15;          // A: rows t/16, t/16+16, .. (4 rows)
  const int bslot = t >> 7, bn = t & 127;         // B: k-slots t/128, +2, .. (8 slots)
  const float* aSrc = Ab + (long long)(m0 + ai) * lda + aslot * 2;
  const float* bSrc = Bb + (long long)(bslot * 2) * ldb + n0 + bn;

  v8f acc00 = {0.f,0.f,0.f,0.f,0.f,0.f,0.f,0.f};
  v8f acc01 = acc00, acc10 = acc00, acc11 = acc00;

  // stage k=0 into buffer 0
#pragma unroll
  for (int q = 0; q < 4; q++) {
    const float2 f = *(const float2*)(aSrc + (long long)q * 16 * lda);
    sA[0][(ai + q * 16) * 17 + aslot] = pack_bf16(f.x, f.y);
  }
#pragma unroll
  for (int q = 0; q < 8; q++) {
    const float* s = bSrc + (long long)(q * 4) * ldb;   // k advances 2 slots (=4 rows) per q
    sB[0][bn * 17 + bslot + q * 2] = pack_bf16(s[0], s[ldb]);
  }
  __syncthreads();

  int buf = 0;
  for (int k0 = 0; k0 < K; k0 += TK) {
    if (k0 + TK < K) {
      const float* aS = aSrc + k0 + TK;
      const float* bS = bSrc + (long long)(k0 + TK) * ldb;
#pragma unroll
      for (int q = 0; q < 4; q++) {
        const float2 f = *(const float2*)(aS + (long long)q * 16 * lda);
        sA[buf ^ 1][(ai + q * 16) * 17 + aslot] = pack_bf16(f.x, f.y);
      }
#pragma unroll
      for (int q = 0; q < 8; q++) {
        const float* s = bS + (long long)(q * 4) * ldb;
        sB[buf ^ 1][bn * 17 + bslot + q * 2] = pack_bf16(s[0], s[ldb]);
      }
    }
    if (k0 + 2 * TK < K) {
      __builtin_prefetch(aSrc + k0 + 2 * TK, 0, 1);
      __builtin_prefetch(bSrc + (long long)(k0 + 2 * TK) * ldb, 0, 1);
    }

    BF16x16 a0, a1, b0, b1;
    gather_frags(sA[buf], sB[buf], wm, wn, g, r, a0, a1, b0, b1);
    acc00 = __builtin_amdgcn_wmma_f32_16x16x32_bf16(false, a0.v, false, b0.v, (short)0, acc00, false, false);
    acc01 = __builtin_amdgcn_wmma_f32_16x16x32_bf16(false, a0.v, false, b1.v, (short)0, acc01, false, false);
    acc10 = __builtin_amdgcn_wmma_f32_16x16x32_bf16(false, a1.v, false, b0.v, (short)0, acc10, false, false);
    acc11 = __builtin_amdgcn_wmma_f32_16x16x32_bf16(false, a1.v, false, b1.v, (short)0, acc11, false, false);

    __syncthreads();
    buf ^= 1;
  }

  epilogue_store(acc00, acc01, acc10, acc11, biasb, Cb, ldc,
                 m0, n0, wm, wn, g, r, 1 << 30, 1 << 30, doGelu, doAccum, alpha);
}

// =====================================================================
// EDGE GEMM: fully guarded, optional transposes. Used only for the three
// small edge-shaped GEMMs (N=18 / M=18 / trans cases).
// =====================================================================
__global__ __launch_bounds__(256)
void gemm_bf16_edge(const float* __restrict__ Ag, int lda, long long strA,
                    const float* __restrict__ Bg, int ldb, long long strB,
                    const float* __restrict__ bias, long long strBias,
                    float* __restrict__ Cg, int ldc, long long strC,
                    int M, int N, int K,
                    int aT, int bT, int doGelu, int doAccum, float alpha)
{
  __shared__ unsigned int sA[2][TM * 17];
  __shared__ unsigned int sB[2][TN * 17];

  const float* Ab = Ag + (long long)blockIdx.z * strA;
  const float* Bb = Bg + (long long)blockIdx.z * strB;
  const float* biasb = bias ? (bias + (long long)blockIdx.z * strBias) : nullptr;
  float* Cb = Cg + (long long)blockIdx.z * strC;

  const int m0 = blockIdx.y * TM;
  const int n0 = blockIdx.x * TN;
  const int t = threadIdx.x;
  const int lane = t & 31;
  const int wave = t >> 5;
  const int wm = (wave & 1) * 32;
  const int wn = (wave >> 1) * 32;
  const int g = lane >> 4;
  const int r = lane & 15;

  v8f acc00 = {0.f,0.f,0.f,0.f,0.f,0.f,0.f,0.f};
  v8f acc01 = acc00, acc10 = acc00, acc11 = acc00;

  int buf = 0;
  for (int k0 = 0; k0 < K; k0 += TK) {
    for (int p = t; p < TM * 16; p += 256) {
      int i = p >> 4, slot = p & 15;
      int gm = m0 + i, gk = k0 + slot * 2;
      float f0 = 0.f, f1 = 0.f;
      if (gm < M) {
        if (!aT) {
          if (gk < K)     f0 = Ab[(long long)gm * lda + gk];
          if (gk + 1 < K) f1 = Ab[(long long)gm * lda + gk + 1];
        } else {
          if (gk < K)     f0 = Ab[(long long)gk * lda + gm];
          if (gk + 1 < K) f1 = Ab[(long long)(gk + 1) * lda + gm];
        }
      }
      sA[buf][i * 17 + slot] = pack_bf16(f0, f1);
    }
    for (int p = t; p < TN * 16; p += 256) {
      int slot = p >> 7, n = p & 127;
      int gn = n0 + n, gk = k0 + slot * 2;
      float f0 = 0.f, f1 = 0.f;
      if (gn < N) {
        if (!bT) {
          if (gk < K)     f0 = Bb[(long long)gk * ldb + gn];
          if (gk + 1 < K) f1 = Bb[(long long)(gk + 1) * ldb + gn];
        } else {
          if (gk < K)     f0 = Bb[(long long)gn * ldb + gk];
          if (gk + 1 < K) f1 = Bb[(long long)gn * ldb + gk + 1];
        }
      }
      sB[buf][n * 17 + slot] = pack_bf16(f0, f1);
    }
    __syncthreads();

    BF16x16 a0, a1, b0, b1;
    gather_frags(sA[buf], sB[buf], wm, wn, g, r, a0, a1, b0, b1);
    acc00 = __builtin_amdgcn_wmma_f32_16x16x32_bf16(false, a0.v, false, b0.v, (short)0, acc00, false, false);
    acc01 = __builtin_amdgcn_wmma_f32_16x16x32_bf16(false, a0.v, false, b1.v, (short)0, acc01, false, false);
    acc10 = __builtin_amdgcn_wmma_f32_16x16x32_bf16(false, a1.v, false, b0.v, (short)0, acc10, false, false);
    acc11 = __builtin_amdgcn_wmma_f32_16x16x32_bf16(false, a1.v, false, b1.v, (short)0, acc11, false, false);

    __syncthreads();
    buf ^= 1;
  }

  epilogue_store(acc00, acc01, acc10, acc11, biasb, Cb, ldc,
                 m0, n0, wm, wn, g, r, M, N, doGelu, doAccum, alpha);
}

// ---------------- pooling: h0 = [mean || max] over masked L ----------------
__global__ void pool_kernel(const float* __restrict__ hidden,
                            const int* __restrict__ am, float* __restrict__ h0) {
  int b = blockIdx.x;
  for (int h = threadIdx.x; h < H_; h += blockDim.x) {
    float s = 0.f, mx = -3.402823466e38f;
    int cnt = 0;
    for (int l = 0; l < L_; l++) {
      if (am[b * L_ + l] != 0) {
        float v = hidden[((long long)b * L_ + l) * H_ + h];
        s += v; mx = fmaxf(mx, v); cnt++;
      }
    }
    float len = (float)(cnt > 0 ? cnt : 1);
    h0[(long long)b * Dm + h]      = s / len;
    h0[(long long)b * Dm + H_ + h] = mx;
  }
}

// ---------------- LayerNorm over last dim (rows = b*E + e) ----------------
__global__ void ln_kernel(float* __restrict__ X, const float* __restrict__ g,
                          const float* __restrict__ be, int E) {
  int row = blockIdx.x;
  int e = row % E;
  float* x = X + (long long)row * Dm;
  __shared__ float red[256];
  int t = threadIdx.x;
  float s = 0.f;
  for (int d = t; d < Dm; d += 256) s += x[d];
  red[t] = s; __syncthreads();
  for (int st = 128; st > 0; st >>= 1) { if (t < st) red[t] += red[t + st]; __syncthreads(); }
  float mu = red[0] / (float)Dm; __syncthreads();
  float s2 = 0.f;
  for (int d = t; d < Dm; d += 256) { float dd = x[d] - mu; s2 += dd * dd; }
  red[t] = s2; __syncthreads();
  for (int st = 128; st > 0; st >>= 1) { if (t < st) red[t] += red[t + st]; __syncthreads(); }
  float inv = rsqrtf(red[0] / (float)Dm + EPS_LN); __syncthreads();
  for (int d = t; d < Dm; d += 256)
    x[d] = (x[d] - mu) * inv * g[e * Dm + d] + be[e * Dm + d];
}

// ---------------- small gate head: softmax(G @ W2 + b2) ----------------
__global__ void gate_kernel(const float* __restrict__ G, const float* __restrict__ W2,
                            const float* __restrict__ b2, float* __restrict__ w, int nout) {
  int b = blockIdx.x, t = threadIdx.x;
  __shared__ float red[256];
  __shared__ float lg[8];
  for (int j = 0; j < nout; j++) {
    float s = 0.f;
    for (int d = t; d < Dm; d += 256) s += G[(long long)b * Dm + d] * W2[d * nout + j];
    red[t] = s; __syncthreads();
    for (int st = 128; st > 0; st >>= 1) { if (t < st) red[t] += red[t + st]; __syncthreads(); }
    if (t == 0) lg[j] = red[0] + b2[j];
    __syncthreads();
  }
  if (t == 0) {
    float mx = -3.402823466e38f;
    for (int j = 0; j < nout; j++) mx = fmaxf(mx, lg[j]);
    float sum = 0.f;
    for (int j = 0; j < nout; j++) { lg[j] = expf(lg[j] - mx); sum += lg[j]; }
    for (int j = 0; j < nout; j++) w[b * nout + j] = lg[j] / sum;
  }
}

// ---------------- PLE expert combination ----------------
__global__ void combine_kernel(const float* __restrict__ Ea, const float* __restrict__ Er,
                               const float* __restrict__ Es, const float* __restrict__ wa,
                               const float* __restrict__ wr, const float* __restrict__ wsg,
                               float* __restrict__ oa, float* __restrict__ orr,
                               float* __restrict__ os) {
  int i = blockIdx.x * 256 + threadIdx.x;
  if (i >= B_ * Dm) return;
  int b = i / Dm, d = i % Dm;
  long long base = (long long)b * 2 * Dm;
  float ea0 = Ea[base + d], ea1 = Ea[base + Dm + d];
  float er0 = Er[base + d], er1 = Er[base + Dm + d];
  float es0 = Es[base + d], es1 = Es[base + Dm + d];
  const float* a4 = wa + b * 4; const float* r4 = wr + b * 4; const float* s6 = wsg + b * 6;
  oa[i]  = a4[0] * ea0 + a4[1] * ea1 + a4[2] * es0 + a4[3] * es1;
  orr[i] = r4[0] * er0 + r4[1] * er1 + r4[2] * es0 + r4[3] * es1;
  os[i]  = s6[0] * ea0 + s6[1] * ea1 + s6[2] * er0 + s6[3] * er1 + s6[4] * es0 + s6[5] * es1;
}

// ---------------- softmax over sequence dim (per b,a) ----------------
__global__ void attn_softmax_kernel(float* __restrict__ S, const int* __restrict__ am) {
  int ba = blockIdx.x;
  int b = ba / A_, a = ba % A_;
  int t = threadIdx.x;
  __shared__ float red[256];
  float mx = -3.402823466e38f;
  for (int l = t; l < L_; l += 256) {
    float v = (am[b * L_ + l] != 0) ? S[((long long)b * L_ + l) * A_ + a] : -3.402823466e38f;
    mx = fmaxf(mx, v);
  }
  red[t] = mx; __syncthreads();
  for (int st = 128; st > 0; st >>= 1) { if (t < st) red[t] = fmaxf(red[t], red[t + st]); __syncthreads(); }
  mx = red[0]; __syncthreads();
  float sum = 0.f;
  for (int l = t; l < L_; l += 256) {
    long long idx = ((long long)b * L_ + l) * A_ + a;
    float v = (am[b * L_ + l] != 0) ? S[idx] : -3.402823466e38f;
    float e = expf(v - mx);
    S[idx] = e; sum += e;
  }
  red[t] = sum; __syncthreads();
  for (int st = 128; st > 0; st >>= 1) { if (t < st) red[t] += red[t + st]; __syncthreads(); }
  float inv = 1.f / red[0]; __syncthreads();
  for (int l = t; l < L_; l += 256) S[((long long)b * L_ + l) * A_ + a] *= inv;
}

// ---------------- ACSA logits: z=[m_proj||ha] @ acsa_W + b ----------------
__global__ void acsa_kernel(const float* __restrict__ mproj, const float* __restrict__ ha,
                            const float* __restrict__ Wc, const float* __restrict__ bc,
                            float* __restrict__ outL) {
  int ba = blockIdx.x, b = ba / A_, t = threadIdx.x;
  __shared__ float red[256];
  const float* mp = mproj + (long long)ba * Dm;
  const float* hb = ha + (long long)b * Dm;
  for (int c = 0; c < 3; c++) {
    float s = 0.f;
    for (int d = t; d < Dm; d += 256) s += mp[d] * Wc[d * 3 + c];
    for (int d = t; d < Dm; d += 256) s += hb[d] * Wc[(Dm + d) * 3 + c];
    red[t] = s; __syncthreads();
    for (int st = 128; st > 0; st >>= 1) { if (t < st) red[t] += red[t + st]; __syncthreads(); }
    if (t == 0) outL[ba * 3 + c] = red[0] + bc[c];
    __syncthreads();
  }
}

// ---------------- broadcast ha over aspects ----------------
__global__ void bcast_kernel(const float* __restrict__ ha, float* __restrict__ harep) {
  int i = blockIdx.x * 256 + threadIdx.x;
  if (i >= B_ * A_ * Dm) return;
  int b = i / (A_ * Dm);
  int d = i % Dm;
  harep[i] = ha[(long long)b * Dm + d];
}

// ---------------- rowdot: out[r] = X[r,:] . w + b ----------------
__global__ void rowdot_kernel(const float* __restrict__ X, const float* __restrict__ w,
                              const float* __restrict__ bias, float* __restrict__ out, int Dlen) {
  int r = blockIdx.x, t = threadIdx.x;
  __shared__ float red[256];
  float s = 0.f;
  for (int d = t; d < Dlen; d += 256) s += X[(long long)r * Dlen + d] * w[d];
  red[t] = s; __syncthreads();
  for (int st = 128; st > 0; st >>= 1) { if (t < st) red[t] += red[t + st]; __syncthreads(); }
  if (t == 0) out[r] = red[0] + (bias ? bias[0] : 0.f);
}

// ---------------- rp sigmoid ----------------
__global__ void rp_kernel(const float* __restrict__ raw, float* __restrict__ rpn) {
  int b = threadIdx.x;
  if (b < B_) rpn[b] = 1.f / (1.f + expf(-raw[b]));
}

// ---------------- per-(b,a) CE terms ----------------
__global__ void nll_kernel(const float* __restrict__ logits, const int* __restrict__ labels,
                           float* __restrict__ nll, float* __restrict__ valid) {
  int i = blockIdx.x * 256 + threadIdx.x;
  if (i >= B_ * A_) return;
  const float* lg = logits + i * 3;
  float m = fmaxf(lg[0], fmaxf(lg[1], lg[2]));
  float lse = m + logf(expf(lg[0] - m) + expf(lg[1] - m) + expf(lg[2] - m));
  int lab = labels[i];
  int cl = lab < 0 ? 0 : (lab > 2 ? 2 : lab);
  float v = (lab != -100) ? 1.f : 0.f;
  nll[i] = v * (lse - lg[cl]);
  valid[i] = v;
}

// ---------------- per-b consistency term ----------------
__global__ void lc_kernel(const float* __restrict__ logits, const float* __restrict__ wpre,
                          const float* __restrict__ stars, const float* __restrict__ cscale,
                          const float* __restrict__ cbias, float* __restrict__ lcb) {
  int b = blockIdx.x, a = threadIdx.x;
  __shared__ float ssi[32], sw[32];
  float si = 0.f, wp = -3.402823466e38f;
  if (a < A_) {
    const float* lg = logits + (b * A_ + a) * 3;
    float m = fmaxf(lg[0], fmaxf(lg[1], lg[2]));
    float e0 = expf(lg[0] - m), e1 = expf(lg[1] - m), e2 = expf(lg[2] - m);
    si = (e2 - e0) / (e0 + e1 + e2);
    wp = wpre[b * A_ + a];
  }
  ssi[a] = si; sw[a] = wp;
  __syncthreads();
  if (a == 0) {
    float mx = -3.402823466e38f;
    for (int i = 0; i < A_; i++) mx = fmaxf(mx, sw[i]);
    float sum = 0.f;
    for (int i = 0; i < A_; i++) { sw[i] = expf(sw[i] - mx); sum += sw[i]; }
    float agg = 0.f;
    for (int i = 0; i < A_; i++) agg += (sw[i] / sum) * ssi[i];
    float r = 1.f / (1.f + expf(-(cscale[0] * agg + cbias[0])));
    float d = r - stars[b];
    float ad = fabsf(d);
    lcb[b] = ad < 1.f ? 0.5f * d * d : ad - 0.5f;
  }
}

// ---------------- final reduction: la, lr, lc, total ----------------
__global__ void final_kernel(const float* __restrict__ nll, const float* __restrict__ valid,
                             const float* __restrict__ lcb, const float* __restrict__ rpn,
                             const float* __restrict__ stars, float* __restrict__ outs) {
  __shared__ float r1[256], r2[256];
  int t = threadIdx.x;
  float sn = 0.f, sv = 0.f;
  for (int i = t; i < B_ * A_; i += 256) { sn += nll[i]; sv += valid[i]; }
  r1[t] = sn; r2[t] = sv; __syncthreads();
  for (int st = 128; st > 0; st >>= 1) {
    if (t < st) { r1[t] += r1[t + st]; r2[t] += r2[t + st]; }
    __syncthreads();
  }
  if (t == 0) {
    float la = r1[0] / fmaxf(r2[0], 1.f);
    float lr = 0.f, lc = 0.f;
    for (int b = 0; b < B_; b++) { lr += fabsf(rpn[b] - stars[b]); lc += lcb[b]; }
    lr /= (float)B_; lc /= (float)B_;
    float total = la + 2.0f * lr + 0.2f * lc;
    outs[0] = total; outs[1] = la; outs[2] = lr; outs[3] = lc;
  }
}

// =====================================================================
// Host-side orchestration
// =====================================================================
static inline void gemm(hipStream_t s,
                        const float* Ap, int lda, long long strA,
                        const float* Bp, int ldb, long long strB,
                        const float* bias, long long strBias,
                        float* Cp, int ldc, long long strC,
                        int M, int N, int K, int batch,
                        int aT, int bT, int gelu, int accum, float alpha) {
  dim3 g((N + TN - 1) / TN, (M + TM - 1) / TM, batch);
  const bool fast = (!aT) && (!bT) && (M % TM == 0) && (N % TN == 0) && (K % TK == 0);
  if (fast) {
    hipLaunchKernelGGL(gemm_bf16_fast, g, dim3(256), 0, s,
                       Ap, lda, strA, Bp, ldb, strB, bias, strBias,
                       Cp, ldc, strC, K, gelu, accum, alpha);
  } else {
    hipLaunchKernelGGL(gemm_bf16_edge, g, dim3(256), 0, s,
                       Ap, lda, strA, Bp, ldb, strB, bias, strBias,
                       Cp, ldc, strC, M, N, K, aT, bT, gelu, accum, alpha);
  }
}

extern "C" void kernel_launch(void* const* d_in, const int* in_sizes, int n_in,
                              void* d_out, int out_size, void* d_ws, size_t ws_size,
                              hipStream_t stream) {
  (void)in_sizes; (void)n_in; (void)out_size; (void)ws_size;
  // ---- input pointers (setup_inputs dict order; params flattened with
  //      sorted dict keys, layers list in order) ----
  const float* hidden = (const float*)d_in[0];
  const int*   am     = (const int*)d_in[1];
  const int*   labels = (const int*)d_in[2];
  const float* stars  = (const float*)d_in[3];
  const float* acsaW  = (const float*)d_in[4];
  const float* acsab  = (const float*)d_in[5];
  const float* aq     = (const float*)d_in[6];
  const float* cbias  = (const float*)d_in[7];
  const float* cscale = (const float*)d_in[8];
  const float* gW1    = (const float*)d_in[9];
  const float* gB1    = (const float*)d_in[10];
  const float* gW2    = (const float*)d_in[11];
  const float* gB2    = (const float*)d_in[12];
  // layer leaves, sorted key order:
  // 0 aW 1 ab 2 abe 3 ag 4 ga1 5 ga2 6 gab1 7 gab2 8 gr1 9 gr2 10 grb1 11 grb2
  // 12 gs1 13 gs2 14 gsb1 15 gsb2 16 rW 17 rb 18 rbe 19 rg 20 sW 21 sb 22 sbe 23 sg
  const float* lp[2][24];
  for (int l = 0; l < 2; l++)
    for (int j = 0; j < 24; j++) lp[l][j] = (const float*)d_in[13 + l * 24 + j];
  const float* projW = (const float*)d_in[61];
  const float* projB = (const float*)d_in[62];
  const float* rpW1  = (const float*)d_in[63];
  const float* rpB1  = (const float*)d_in[64];
  const float* rpW2  = (const float*)d_in[65];
  const float* rpB2  = (const float*)d_in[66];
  const float* wq    = (const float*)d_in[67];

  float* out = (float*)d_out;
  float* outLogits = out;                 // [B,A,3]
  float* outRp     = out + B_ * A_ * 3;   // [B]
  float* outScal   = outRp + B_;          // total, la, lr, lc

  // ---- workspace layout ----
  float* W = (float*)d_ws;
  size_t o = 0;
  float* h0    = W + o; o += (size_t)B_ * Dm;
  float* hn[2][3];
  for (int l = 0; l < 2; l++)
    for (int j = 0; j < 3; j++) { hn[l][j] = W + o; o += (size_t)B_ * Dm; }
  float* Ea = W + o; o += (size_t)B_ * 2 * Dm;
  float* Er = W + o; o += (size_t)B_ * 2 * Dm;
  float* Es = W + o; o += (size_t)B_ * 2 * Dm;
  float* Gh = W + o; o += (size_t)B_ * Dm;
  float* wa_ = W + o; o += B_ * 4;
  float* wr_ = W + o; o += B_ * 4;
  float* wsg_ = W + o; o += B_ * 6;
  float* wqa = W + o; o += (size_t)H_ * A_;
  float* scores = W + o; o += (size_t)B_ * L_ * A_;
  float* mbuf   = W + o; o += (size_t)B_ * A_ * H_;
  float* mproj  = W + o; o += (size_t)B_ * A_ * Dm;
  float* harep  = W + o; o += (size_t)B_ * A_ * Dm;
  float* G1     = W + o; o += (size_t)B_ * A_ * (Dm / 2);
  float* hrg    = W + o; o += (size_t)B_ * (Dm / 2);
  float* wpre   = W + o; o += B_ * A_;
  float* rpraw  = W + o; o += B_;
  float* nllb   = W + o; o += B_ * A_;
  float* validb = W + o; o += B_ * A_;
  float* lcb    = W + o; o += B_;

  // ---- 1. sentence pooling ----
  hipLaunchKernelGGL(pool_kernel, dim3(B_), dim3(256), 0, stream, hidden, am, h0);

  // ---- 2. PLE layers ----
  const float* ia = h0; const float* ir = h0; const float* is = h0;
  for (int l = 0; l < 2; l++) {
    const float** P = lp[l];
    // experts (batched over E=2): [64,1536]x[1536,1536], bias, GELU
    gemm(stream, ia, Dm, 0, P[0], Dm, (long long)Dm * Dm, P[1], Dm,
         Ea, 2 * Dm, Dm, B_, Dm, Dm, 2, 0, 0, 1, 0, 1.f);
    gemm(stream, ir, Dm, 0, P[16], Dm, (long long)Dm * Dm, P[17], Dm,
         Er, 2 * Dm, Dm, B_, Dm, Dm, 2, 0, 0, 1, 0, 1.f);
    gemm(stream, is, Dm, 0, P[20], Dm, (long long)Dm * Dm, P[21], Dm,
         Es, 2 * Dm, Dm, B_, Dm, Dm, 2, 0, 0, 1, 0, 1.f);
    hipLaunchKernelGGL(ln_kernel, dim3(B_ * 2), dim3(256), 0, stream, Ea, P[3], P[2], 2);
    hipLaunchKernelGGL(ln_kernel, dim3(B_ * 2), dim3(256), 0, stream, Er, P[19], P[18], 2);
    hipLaunchKernelGGL(ln_kernel, dim3(B_ * 2), dim3(256), 0, stream, Es, P[23], P[22], 2);
    // gates: Gh = GELU(x @ g1 + b1); w = softmax(Gh @ g2 + b2)
    gemm(stream, ia, Dm, 0, P[4], Dm, 0, P[6], 0, Gh, Dm, 0, B_, Dm, Dm, 1, 0, 0, 1, 0, 1.f);
    hipLaunchKernelGGL(gate_kernel, dim3(B_), dim3(256), 0, stream, Gh, P[5], P[7], wa_, 4);
    gemm(stream, ir, Dm, 0, P[8], Dm, 0, P[10], 0, Gh, Dm, 0, B_, Dm, Dm, 1, 0, 0, 1, 0, 1.f);
    hipLaunchKernelGGL(gate_kernel, dim3(B_), dim3(256), 0, stream, Gh, P[9], P[11], wr_, 4);
    gemm(stream, is, Dm, 0, P[12], Dm, 0, P[14], 0, Gh, Dm, 0, B_, Dm, Dm, 1, 0, 0, 1, 0, 1.f);
    hipLaunchKernelGGL(gate_kernel, dim3(B_), dim3(256), 0, stream, Gh, P[13], P[15], wsg_, 6);
    // combine
    hipLaunchKernelGGL(combine_kernel, dim3((B_ * Dm + 255) / 256), dim3(256), 0, stream,
                       Ea, Er, Es, wa_, wr_, wsg_, hn[l][0], hn[l][1], hn[l][2]);
    ia = hn[l][0]; ir = hn[l][1]; is = hn[l][2];
  }
  const float* haF = ia;   // final ha
  const float* hrF = ir;   // final hr

  // ---- 3. aspect attention pooling ----
  // wqa = (wq @ aq^T) / sqrt(H)   [768 x 18]
  gemm(stream, wq, H_, 0, aq, H_, 0, nullptr, 0, wqa, A_, 0,
       H_, A_, H_, 1, 0, 1, 0, 0, 1.0f / sqrtf((float)H_));
  // scores = hidden2d @ wqa       [32768 x 18]
  gemm(stream, hidden, H_, 0, wqa, A_, 0, nullptr, 0, scores, A_, 0,
       B_ * L_, A_, H_, 1, 0, 0, 0, 0, 1.f);
  hipLaunchKernelGGL(attn_softmax_kernel, dim3(B_ * A_), dim3(256), 0, stream, scores, am);
  // m[b] = attn[b]^T @ hidden[b]  (batched, trans-A)   [18 x 768] per b
  gemm(stream, scores, A_, (long long)L_ * A_, hidden, H_, (long long)L_ * H_,
       nullptr, 0, mbuf, H_, (long long)A_ * H_, A_, H_, L_, B_, 1, 0, 0, 0, 1.f);
  // m_proj = m @ proj_W + proj_b  [1152 x 1536]
  gemm(stream, mbuf, H_, 0, projW, Dm, 0, projB, 0, mproj, Dm, 0,
       B_ * A_, Dm, H_, 1, 0, 0, 0, 0, 1.f);

  // ---- 4. heads ----
  hipLaunchKernelGGL(acsa_kernel, dim3(B_ * A_), dim3(256), 0, stream,
                     mproj, haF, acsaW, acsab, outLogits);
  hipLaunchKernelGGL(bcast_kernel, dim3((B_ * A_ * Dm + 255) / 256), dim3(256), 0, stream,
                     haF, harep);
  // gate hidden: split-K over z = [m_proj || ha_rep]
  gemm(stream, mproj, Dm, 0, gW1, Dm / 2, 0, nullptr, 0, G1, Dm / 2, 0,
       B_ * A_, Dm / 2, Dm, 1, 0, 0, 0, 0, 1.f);
  gemm(stream, harep, Dm, 0, gW1 + (size_t)Dm * (Dm / 2), Dm / 2, 0, gB1, 0,
       G1, Dm / 2, 0, B_ * A_, Dm / 2, Dm, 1, 0, 0, 1, 1, 1.f);
  hipLaunchKernelGGL(rowdot_kernel, dim3(B_ * A_), dim3(256), 0, stream,
                     G1, gW2, gB2, wpre, Dm / 2);
  // rp head
  gemm(stream, hrF, Dm, 0, rpW1, Dm / 2, 0, rpB1, 0, hrg, Dm / 2, 0,
       B_, Dm / 2, Dm, 1, 0, 0, 1, 0, 1.f);
  hipLaunchKernelGGL(rowdot_kernel, dim3(B_), dim3(256), 0, stream, hrg, rpW2, rpB2, rpraw, Dm / 2);
  hipLaunchKernelGGL(rp_kernel, dim3(1), dim3(64), 0, stream, rpraw, outRp);

  // ---- 5. losses ----
  hipLaunchKernelGGL(nll_kernel, dim3((B_ * A_ + 255) / 256), dim3(256), 0, stream,
                     outLogits, labels, nllb, validb);
  hipLaunchKernelGGL(lc_kernel, dim3(B_), dim3(32), 0, stream,
                     outLogits, wpre, stars, cscale, cbias, lcb);
  hipLaunchKernelGGL(final_kernel, dim3(1), dim3(256), 0, stream,
                     nllb, validb, lcb, outRp, stars, outScal);
}
